// TuckerGroupLinear_19859928776889
// MI455X (gfx1250) — matrix operand order repaired and checked by reading, not compiled, not run on hardware
//
#include <hip/hip_runtime.h>
#include <hip/hip_bf16.h>

// Tucker grouped linear for MI455X (gfx1250, wave32):
//   stage 0: U_inT = transpose(U_in)      (one-time 1MB layout fix -> enables async fills)
//   stage 1: x_low = x @ U_in             (WMMA bf16, async-to-LDS double-buffered B)
//   stage 2: y[n]  = W_low[e[n]] @ x_low  (VALU gather-dot, W_low L2-resident)
//   stage 3: out   = y @ U_out.T          (WMMA bf16, TDM tensor_load_to_lds B tile)

#define N_TOK 8192
#define KDIM  4096
#define RDIM  128
#define RUDIM 128
#define ODIM  4096

typedef __attribute__((ext_vector_type(16))) __bf16 bf16x16;
typedef __attribute__((ext_vector_type(8)))  float  f32x8;
typedef __attribute__((ext_vector_type(4))) unsigned int uint32x4;
typedef __attribute__((ext_vector_type(8))) unsigned int uint32x8;

union FragBF { unsigned int u[8]; uint4 q[2]; bf16x16 v; };

__device__ __forceinline__ float bflo(unsigned int p) { return __uint_as_float(p << 16); }
__device__ __forceinline__ float bfhi(unsigned int p) { return __uint_as_float(p & 0xFFFF0000u); }
__device__ __forceinline__ unsigned short f2bf(float f) {
  unsigned int u = __float_as_uint(f);
  return (unsigned short)((u + 0x7FFFu + ((u >> 16) & 1u)) >> 16); // RNE
}
__device__ __forceinline__ unsigned int f2bf_pk(float lo, float hi) {
  return (unsigned int)f2bf(lo) | ((unsigned int)f2bf(hi) << 16);
}

// Async cache->LDS b128 copy (gfx1250 ASYNCcnt path); per-lane addresses.
__device__ __forceinline__ void async_b128(unsigned int lds_off, const void* gptr) {
  unsigned long long g = (unsigned long long)(uintptr_t)gptr;
  asm volatile("global_load_async_to_lds_b128 %0, %1, off"
               :: "v"(lds_off), "v"(g) : "memory");
}
__device__ __forceinline__ void wait_async0() {
  asm volatile("s_wait_asynccnt 0" ::: "memory");
}
__device__ __forceinline__ unsigned int lds_off_u32(const void* p) {
  return (unsigned int)(uintptr_t)p;  // generic LDS pointer: low 32 bits = LDS offset
}

// ---------------------------------------------------------------------------
// Stage 0: U_inT[128,4096] = U_in[4096,128]^T  (coalesced 32x32 LDS tiles)
// ---------------------------------------------------------------------------
__global__ __launch_bounds__(256) void k_transpose(
    const unsigned short* __restrict__ in, unsigned short* __restrict__ out)
{
  __shared__ unsigned short t[32][33];
  const int kx = blockIdx.x * 32;
  const int nx = blockIdx.y * 32;
  const int lx = threadIdx.x & 31;
  const int ly = threadIdx.x >> 5;  // 0..7
#pragma unroll
  for (int i = 0; i < 32; i += 8)
    t[ly + i][lx] = in[(size_t)(kx + ly + i) * RDIM + nx + lx];
  __syncthreads();
#pragma unroll
  for (int i = 0; i < 32; i += 8)
    out[(size_t)(nx + ly + i) * KDIM + kx + lx] = t[lx][ly + i];
}

// ---------------------------------------------------------------------------
// Stage 1: x_low[8192,128] = x[8192,4096] @ U_in[4096,128].
// 128 threads = 4 waves; block tile 64(M) x 128(N); K chunk 32; async double-
// buffered LDS B tile.
// ---------------------------------------------------------------------------
__global__ __launch_bounds__(128) void k_gemm_down(
    const unsigned short* __restrict__ A,     // x
    const unsigned short* __restrict__ BT,    // U_inT [128,4096]
    unsigned short* __restrict__ C)           // x_low
{
  // [buf][n][k], row stride 40 elems (80B; b128 frags at n*40+16*hi are aligned)
  __shared__ __align__(16) unsigned short Bt[2][128 * 40];

  const int tid  = threadIdx.x;
  const int wave = tid >> 5;
  const int lane = tid & 31;
  const int l16  = lane & 15;
  const int hi   = lane >> 4;
  const int mBase = blockIdx.x * 64 + wave * 16;
  const int aRow  = mBase + l16;

  const uint4* A4 = (const uint4*)A;   // 512 uint4 per 4096-elem row
  const uint4* B4 = (const uint4*)BT;  // 512 uint4 per 4096-elem row

  f32x8 acc[8];
#pragma unroll
  for (int i = 0; i < 8; ++i)
#pragma unroll
    for (int j = 0; j < 8; ++j) acc[i][j] = 0.0f;

  // Async fill of one 128(n) x 32(k) chunk: 512 b128, 4 per thread.
  auto fill = [&](int buf, int kc) {
#pragma unroll
    for (int i = 0; i < 4; ++i) {
      int f = tid + 128 * i;
      int r = f >> 2;
      int c = f & 3;
      async_b128(lds_off_u32(&Bt[buf][r * 40 + c * 8]),
                 &B4[(size_t)r * 512 + (kc >> 3) + c]);
    }
  };

  // One k-chunk: A frag (2x b128) + 8 WMMAs fed by ds_load_b128 pairs.
  auto compute = [&](int buf, int kc) {
    FragBF a;
    a.q[0] = A4[(size_t)aRow * 512 + (kc >> 3) + 2 * hi];
    a.q[1] = A4[(size_t)aRow * 512 + (kc >> 3) + 2 * hi + 1];

    const uint4* Bl = (const uint4*)&Bt[buf][0];
    const int fb = 2 * hi;            // uint4 index of frag base within row (5 per row)
#pragma unroll
    for (int nb = 0; nb < 8; ++nb) {
      FragBF b;
      int n = nb * 16 + l16;
      b.q[0] = Bl[n * 5 + fb];
      b.q[1] = Bl[n * 5 + fb + 1];
      acc[nb] = __builtin_amdgcn_wmma_f32_16x16x32_bf16(
          false, a.v, false, b.v, (short)0, acc[nb], false, false);
    }
  };

  fill(0, 0);
  wait_async0();
  __syncthreads();

  int cur = 0;
#pragma unroll 1
  for (int kc = 0; kc < KDIM - 32; kc += 32) {   // steady state: branch-free body
    fill(cur ^ 1, kc + 32);
    __builtin_prefetch(&A[(size_t)aRow * KDIM + kc + 64], 0, 3);
    compute(cur, kc);
    wait_async0();
    __syncthreads();
    cur ^= 1;
  }
  compute(cur, KDIM - 32);                       // peeled last chunk

#pragma unroll
  for (int nb = 0; nb < 8; ++nb)
#pragma unroll
    for (int v = 0; v < 8; ++v)
      C[(size_t)(mBase + v + 8 * hi) * RDIM + nb * 16 + l16] = f2bf(acc[nb][v]);
}

// ---------------------------------------------------------------------------
// Stage 2: y[n,u] = sum_d x_low[n,d] * W_low[e[n],u,d]. One wave per token.
// ---------------------------------------------------------------------------
__global__ __launch_bounds__(256) void k_expert(
    const unsigned short* __restrict__ Xl,
    const int* __restrict__ eidx,
    const unsigned short* __restrict__ W,
    unsigned short* __restrict__ Y)
{
  __shared__ float xr[8][128];

  const int tid  = threadIdx.x;
  const int wave = tid >> 5;
  const int lane = tid & 31;
  const int n    = blockIdx.x * 8 + wave;

  const uint2* X2 = (const uint2*)Xl;             // 32 uint2 per 128-elem row
  const unsigned int* Wdw = (const unsigned int*)W;

  uint2 p = X2[(size_t)n * 32 + lane];            // global_load_b64
  xr[wave][lane * 4 + 0] = bflo(p.x);
  xr[wave][lane * 4 + 1] = bfhi(p.x);
  xr[wave][lane * 4 + 2] = bflo(p.y);
  xr[wave][lane * 4 + 3] = bfhi(p.y);
  __syncthreads();

  const int e = eidx[n];
  const size_t wBase = (size_t)e * 128 * 64;
  float acc[4] = {0.f, 0.f, 0.f, 0.f};

#pragma unroll 1
  for (int d2 = 0; d2 < 64; ++d2) {
    float xa = xr[wave][2 * d2];
    float xb = xr[wave][2 * d2 + 1];
#pragma unroll
    for (int i = 0; i < 4; ++i) {
      unsigned int wp = Wdw[wBase + (size_t)(lane * 4 + i) * 64 + d2];
      acc[i] += bflo(wp) * xa + bfhi(wp) * xb;
    }
  }

  uint2 r;                                        // 4 contiguous bf16 -> one b64 store
  r.x = f2bf_pk(acc[0], acc[1]);
  r.y = f2bf_pk(acc[2], acc[3]);
  ((uint2*)Y)[(size_t)n * 32 + lane] = r;
}

// ---------------------------------------------------------------------------
// Stage 3: out[8192,4096] = Y[8192,128] @ U_out.T.
// B[u,o] = U_out[o,u]: U_out rows are already K-major. The whole 128x256B
// B tile is fetched by ONE Tensor Data Mover descriptor (TENSORcnt path),
// with TDM pad_interval/pad_amount reproducing the 272B LDS row stride.
// Grid (64, 32); block tile 128(M) x 128(N); K = 128 -> 4 WMMA k-steps.
// ---------------------------------------------------------------------------
__global__ __launch_bounds__(256) void k_gemm_up(
    const unsigned short* __restrict__ Y,
    const unsigned short* __restrict__ Uo,
    unsigned short* __restrict__ Out)
{
  // [n][k], row stride 136 elems = 272B = 64 data dwords + 4 pad dwords
  __shared__ __align__(16) unsigned short Bt[128 * 136];

  const int tid  = threadIdx.x;
  const int wave = tid >> 5;
  const int lane = tid & 31;
  const int l16  = lane & 15;
  const int hi   = lane >> 4;
  const int mBase = blockIdx.x * 128 + wave * 16;
  const int oBase = blockIdx.y * 128;

  const uint4* Y4 = (const uint4*)Y;    // 16 uint4 per 128-elem row

  if (wave == 0) {
    // Tensor DMA descriptor (D#), data_size = 4B (dword units).
    // Tile: dim0 = 64 dwords (one U_out row), dim1 = 128 rows.
    unsigned long long ga =
        (unsigned long long)(uintptr_t)Uo + (unsigned long long)oBase * (RUDIM * 2);
    uint32x4 g0;
    g0.x = 1u;                                            // count=1, user mode
    g0.y = lds_off_u32(&Bt[0]);                           // lds_addr
    g0.z = (unsigned int)ga;                              // global_addr[31:0]
    g0.w = ((unsigned int)(ga >> 32) & 0x01FFFFFFu)       // global_addr[56:32]
         | (2u << 30);                                    // type=2 ("image")
    uint32x8 g1;
    g1.s0 = (2u << 16)        // data_size = 4B
          | (1u << 20)        // pad_enable
          | (5u << 22)        // pad_interval: 64 dwords (one 256B row)
          | (3u << 25);       // pad_amount: 4 dwords (16B)
    g1.s1 = 64u << 16;        // tensor_dim0 = 64 dwords   (bits 79:48)
    g1.s2 = 4096u << 16;      // tensor_dim1 = 4096 rows   (bits 111:80)
    g1.s3 = 64u << 16;        // tile_dim0 = 64 dwords     (bits 127:112)
    g1.s4 = 128u;             // tile_dim1 = 128 rows; tile_dim2 = 0
    g1.s5 = 64u;              // tensor_dim0_stride = 64 dwords (bits 207:160)
    g1.s6 = 0u;               // tensor_dim1_stride = 0 (2D tile)
    g1.s7 = 0u;
    asm volatile("tensor_load_to_lds %0, %1" :: "s"(g0), "s"(g1) : "memory");
    __builtin_amdgcn_s_wait_tensorcnt(0);
  }
  __syncthreads();

  f32x8 acc[8];
#pragma unroll
  for (int i = 0; i < 8; ++i)
#pragma unroll
    for (int j = 0; j < 8; ++j) acc[i][j] = 0.0f;

  const uint4* Bl = (const uint4*)&Bt[0];
#pragma unroll
  for (int kk = 0; kk < 4; ++kk) {
    FragBF a;
    a.q[0] = Y4[(size_t)(mBase + l16) * 16 + kk * 4 + 2 * hi];
    a.q[1] = Y4[(size_t)(mBase + l16) * 16 + kk * 4 + 2 * hi + 1];

    const int fb = kk * 4 + 2 * hi;   // uint4 index of frag base within row (17 per row)
#pragma unroll
    for (int nb = 0; nb < 8; ++nb) {
      FragBF b;
      int n = nb * 16 + l16;
      b.q[0] = Bl[n * 17 + fb];
      b.q[1] = Bl[n * 17 + fb + 1];
      acc[nb] = __builtin_amdgcn_wmma_f32_16x16x32_bf16(
          false, a.v, false, b.v, (short)0, acc[nb], false, false);
    }
  }

#pragma unroll
  for (int nb = 0; nb < 8; ++nb)
#pragma unroll
    for (int v = 0; v < 8; ++v)
      Out[(size_t)(mBase + v + 8 * hi) * ODIM + oBase + nb * 16 + l16] =
          f2bf(acc[nb][v]);
}

// ---------------------------------------------------------------------------
extern "C" void kernel_launch(void* const* d_in, const int* in_sizes, int n_in,
                              void* d_out, int out_size, void* d_ws, size_t ws_size,
                              hipStream_t stream) {
  (void)in_sizes; (void)n_in; (void)out_size; (void)ws_size;

  const unsigned short* x    = (const unsigned short*)d_in[0];  // bf16 [8192,4096]
  const int*            eidx = (const int*)d_in[1];             // i32  [8192]
  const unsigned short* Uin  = (const unsigned short*)d_in[2];  // bf16 [4096,128]
  const unsigned short* Wlow = (const unsigned short*)d_in[3];  // bf16 [64,128,128]
  const unsigned short* Uout = (const unsigned short*)d_in[4];  // bf16 [4096,128]
  unsigned short*       out  = (unsigned short*)d_out;          // bf16 [8192,4096]

  unsigned short* xlow = (unsigned short*)d_ws;                 // bf16 [8192,128]
  unsigned short* y    = xlow + (size_t)N_TOK * RDIM;           // bf16 [8192,128]
  unsigned short* UinT = y + (size_t)N_TOK * RUDIM;             // bf16 [128,4096]

  k_transpose<<<dim3(KDIM / 32, RDIM / 32), 256, 0, stream>>>(Uin, UinT);
  k_gemm_down<<<dim3(N_TOK / 64), 128, 0, stream>>>(x, UinT, xlow);
  k_expert   <<<dim3(N_TOK / 8), 256, 0, stream>>>(xlow, eidx, Wlow, y);
  k_gemm_up  <<<dim3(N_TOK / 128, ODIM / 128), 256, 0, stream>>>(y, Uout, out);
}
